// Model_6330781794327
// MI455X (gfx1250) — matrix operand order repaired and checked by reading, not compiled, not run on hardware
//
#include <hip/hip_runtime.h>
#include <hip/hip_bf16.h>

// ---------------------------------------------------------------------------
// Model constants
// ---------------------------------------------------------------------------
#define MB_  16      // batch
#define MT_  800     // time steps
#define MF_  120     // input features
#define MH_  512     // hidden
#define MC_  5000    // classes
#define MNL_ 100     // decoder steps
#define KC_  5056    // classes padded to a multiple of 64 (K of W_ys GEMM)

typedef __attribute__((ext_vector_type(16))) __bf16 v16bf;
typedef __attribute__((ext_vector_type(8)))  float  v8f;
typedef __hip_bfloat16 bf16;

union FragU { v16bf v; uint4 q[2]; };

// A fragment (16x32 bf16, lane=M row): elements K = koff+{0..7}, koff+{16..23}
// caller passes p = rowptr + koffA (koffA = (lane>>4)*8)
__device__ __forceinline__ v16bf ld_fragA(const bf16* p) {
  FragU f;
  f.q[0] = *reinterpret_cast<const uint4*>(p);
  f.q[1] = *reinterpret_cast<const uint4*>(p + 16);
  return f.v;
}
// B fragment (32x16 bf16, lane=N col): elements K = koff+{0..7}, koff+{16..23}
// caller passes p = rowptr + koffB (koffB = (lane>>4)*8)
__device__ __forceinline__ v16bf ld_fragB(const bf16* p) {
  FragU f;
  f.q[0] = *reinterpret_cast<const uint4*>(p);
  f.q[1] = *reinterpret_cast<const uint4*>(p + 16);
  return f.v;
}
__device__ __forceinline__ v8f wmma_bf16(v16bf a, v16bf b, v8f c) {
  return __builtin_amdgcn_wmma_f32_16x16x32_bf16(false, a, false, b, (short)0, c,
                                                 false, false);
}
__device__ __forceinline__ float sigmoidf_(float x) {
  return 1.0f / (1.0f + __expf(-x));
}

// ---------------------------------------------------------------------------
// Generic WMMA GEMM:  C[M,N] = A[M,K] @ W[N,K]^T (+bias) (+addsrc) (+=C)
// A: bf16 row-major (lda mult of 32), W: bf16 [N,K] row-major (ldw mult of 32),
// K a multiple of 64.  Each wave computes a 16x64 output strip.
// Software-pipelined: fragments consumed by this chunk's WMMAs were loaded in
// the previous iteration; next-chunk loads are issued before the WMMAs so the
// two fragment sets are simultaneously live -> distinct registers -> partial
// s_wait_loadcnt and load/WMMA overlap.
// ---------------------------------------------------------------------------
__global__ void __launch_bounds__(256) k_gemm(
    const bf16* __restrict__ A, int lda,
    const bf16* __restrict__ W, int ldw,
    float* __restrict__ C, long long ldc,
    const float* __restrict__ bias,
    const float* __restrict__ addsrc, long long ldadd,
    int M, int N, int Npad16, int K, int accum)
{
  const int lane = threadIdx.x & 31;
  const int wid  = blockIdx.x * (blockDim.x >> 5) + (threadIdx.x >> 5);
  const int n64  = (Npad16 + 63) >> 6;
  const int mt   = wid / n64;
  if (mt * 16 >= M) return;
  const int nbase = (wid % n64) * 64;

  const int r     = lane & 15;
  const int koffA = (lane >> 4) << 3;
  const int koffB = (lane >> 4) << 3;

  const v8f vz = {0.f,0.f,0.f,0.f,0.f,0.f,0.f,0.f};
  v8f acc[4];
  const bf16* pb[4];
#pragma unroll
  for (int s = 0; s < 4; ++s) {
    acc[s] = vz;
    const int nn   = nbase + s * 16;
    const int nrow = (nn < Npad16) ? nn : 0;   // clamp: keep loads in-bounds
    pb[s] = W + (long long)(nrow + r) * ldw + koffB;
  }
  const bf16* pa = A + (long long)(mt * 16 + r) * lda + koffA;

  // pipeline prologue: fragments for chunk 0
  v16bf aC = ld_fragA(pa);
  v16bf bC[4];
#pragma unroll
  for (int s = 0; s < 4; ++s) bC[s] = ld_fragB(pb[s]);

  for (int kk = 0; kk < K; kk += 64) {
    const int k1 = kk + 32;
    const int k2 = (kk + 64 < K) ? (kk + 64) : 0;   // branchless wrap
    // issue loads for chunk k1 before consuming chunk kk
    v16bf aN = ld_fragA(pa + k1);
    v16bf bN[4];
#pragma unroll
    for (int s = 0; s < 4; ++s) bN[s] = ld_fragB(pb[s] + k1);
    __builtin_prefetch(pb[0] + kk + 512, 0, 3);   // global_prefetch_b8
    __builtin_prefetch(pb[3] + kk + 512, 0, 3);
#pragma unroll
    for (int s = 0; s < 4; ++s) acc[s] = wmma_bf16(aC, bC[s], acc[s]);
    // issue loads for chunk k2 before consuming chunk k1
    aC = ld_fragA(pa + k2);
#pragma unroll
    for (int s = 0; s < 4; ++s) bC[s] = ld_fragB(pb[s] + k2);
#pragma unroll
    for (int s = 0; s < 4; ++s) acc[s] = wmma_bf16(aN, bN[s], acc[s]);
  }

  const int row0 = (lane >> 4) << 3;   // C tile: lanes16-31 hold M rows 8..15
  const int cix  = lane & 15;
#pragma unroll
  for (int s = 0; s < 4; ++s) {
    const int n = nbase + s * 16 + cix;
    if (nbase + s * 16 >= Npad16 || n >= N) continue;
#pragma unroll
    for (int i = 0; i < 8; ++i) {
      const int m = mt * 16 + row0 + i;
      if (m >= M) continue;
      float v = acc[s][i];
      if (bias)   v += bias[n];
      if (addsrc) v += addsrc[(long long)m * ldadd + n];
      float* cp = C + (long long)m * ldc + n;
      if (accum) v += *cp;
      *cp = v;
    }
  }
}

// ---------------------------------------------------------------------------
// Persistent bidirectional LSTM layer. grid = 2 blocks (dir), 512 threads.
// Per step: gates = xproj[t] + h @ Whh^T  (WMMA), then fused cell update.
// Wave w owns h columns [w*32, w*32+32): its 8 accumulator tiles are exactly
// the i/f/g/o gate slices needed for those columns.  h (bf16) and c (f32)
// live in LDS; the rotating B-fragment pipeline wraps across time steps
// (Whh is step-invariant), so chunk-0 weights for step t+1 are already in
// flight while step t's cell update runs.
// ---------------------------------------------------------------------------
__global__ void __launch_bounds__(512) k_lstm(
    const bf16* __restrict__ WhhF, const bf16* __restrict__ WhhB,
    const float* __restrict__ xpF, const float* __restrict__ xpB,
    const int* __restrict__ lens, float* __restrict__ out /* [T][16][1024] */)
{
  const int dir          = blockIdx.x;
  const bf16* Whh        = dir ? WhhB : WhhF;
  const float* xproj     = dir ? xpB  : xpF;
  const int colOff       = dir ? MH_  : 0;

  __shared__ __align__(16) bf16  hbuf[16][MH_];   // 16KB: h state (bf16)
  __shared__ __align__(16) float cbuf[16][MH_];   // 32KB: c state (f32)

  const int tid  = threadIdx.x;
  const int lane = tid & 31;
  const int wave = tid >> 5;

  for (int i = tid; i < 16 * MH_; i += 512) {
    ((bf16*)hbuf)[i]  = __float2bfloat16(0.0f);
    ((float*)cbuf)[i] = 0.0f;
  }
  __syncthreads();

  const int r     = lane & 15;
  const int koffA = (lane >> 4) << 3;
  const int koffB = (lane >> 4) << 3;
  const int cix   = lane & 15;
  const int row0  = (lane >> 4) << 3;

  int mylen[8];
#pragma unroll
  for (int i = 0; i < 8; ++i) mylen[i] = lens[row0 + i];

  // lane-dependent base + uniform per-gate offsets (q = h*4+g)
  const bf16* pbase = Whh + (long long)(wave * 32 + r) * MH_ + koffB;
#define PBOFF(q) ((long long)(((q) & 3) * MH_ + ((q) >> 2) * 16) * MH_)

  const v8f vz = {0.f,0.f,0.f,0.f,0.f,0.f,0.f,0.f};

  // pipeline prologue: B fragments for chunk 0
  v16bf bA[8], bB[8];
#pragma unroll
  for (int q = 0; q < 8; ++q) bA[q] = ld_fragB(pbase + PBOFF(q));

  for (int step = 0; step < MT_; ++step) {
    const int t = dir ? (MT_ - 1 - step) : step;
    v8f acc[8];
#pragma unroll
    for (int q = 0; q < 8; ++q) acc[q] = vz;

    const bf16* pa = &hbuf[r][koffA];
    for (int kk = 0; kk < MH_; kk += 64) {
      const int k1 = kk + 32;
      const int k2 = (kk + 64) & (MH_ - 1);   // wraps to 0 -> next step's chunk 0
#pragma unroll
      for (int q = 0; q < 8; ++q) bB[q] = ld_fragB(pbase + PBOFF(q) + k1);
      v16bf a0 = ld_fragA(pa + kk);
#pragma unroll
      for (int q = 0; q < 8; ++q) acc[q] = wmma_bf16(a0, bA[q], acc[q]);
#pragma unroll
      for (int q = 0; q < 8; ++q) bA[q] = ld_fragB(pbase + PBOFF(q) + k2);
      v16bf a1 = ld_fragA(pa + k1);
#pragma unroll
      for (int q = 0; q < 8; ++q) acc[q] = wmma_bf16(a1, bB[q], acc[q]);
    }
    __syncthreads();   // everyone done reading hbuf for this step

#pragma unroll
    for (int h = 0; h < 2; ++h) {
      const int j = wave * 32 + h * 16 + cix;   // my h column
#pragma unroll
      for (int i = 0; i < 8; ++i) {
        const int m = row0 + i;                 // my batch row
        const float* xr = xproj + ((long long)t * MB_ + m) * 2048 + j;
        float iv = sigmoidf_(acc[h * 4 + 0][i] + xr[0]);
        float fv = sigmoidf_(acc[h * 4 + 1][i] + xr[512]);
        float gv = tanhf    (acc[h * 4 + 2][i] + xr[1024]);
        float ov = sigmoidf_(acc[h * 4 + 3][i] + xr[1536]);
        float cn = fv * cbuf[m][j] + iv * gv;
        float hn = ov * tanhf(cn);
        const bool live = (t < mylen[i]);
        out[((long long)t * MB_ + m) * 1024 + colOff + j] = live ? hn : 0.0f;
        if (live) {                             // masked hold: state untouched
          cbuf[m][j] = cn;
          hbuf[m][j] = __float2bfloat16(hn);
        }
      }
    }
    __syncthreads();   // state updated for next step
  }
#undef PBOFF
}

// ---------------------------------------------------------------------------
// Conversions
// ---------------------------------------------------------------------------
__global__ void k_cvt(const float* __restrict__ src, int sR, int sC,
                      bf16* __restrict__ dst, int dR, int dC)
{
  long long idx = (long long)blockIdx.x * blockDim.x + threadIdx.x;
  if (idx >= (long long)dR * dC) return;
  int rr = (int)(idx / dC), cc = (int)(idx % dC);
  float v = (rr < sR && cc < sC) ? src[(long long)rr * sC + cc] : 0.0f;
  dst[idx] = __float2bfloat16(v);
}

// x [B][T][120] -> bf16 [T*16][128] (K padded to 128)
__global__ void k_cvt_x(const float* __restrict__ x, bf16* __restrict__ dst)
{
  int idx = blockIdx.x * blockDim.x + threadIdx.x;
  if (idx >= MT_ * MB_ * 128) return;
  int rr = idx >> 7, cc = idx & 127;
  int t = rr >> 4, b = rr & 15;
  float v = (cc < MF_) ? x[((long long)b * MT_ + t) * MF_ + cc] : 0.0f;
  dst[idx] = __float2bfloat16(v);
}

// ---------------------------------------------------------------------------
// Decoder element-wise kernels
// ---------------------------------------------------------------------------
// location conv: convout[b][t][f] = sum_k alpha[b][t-50+k] * cw[f][k]
__global__ void k_conv(const float* __restrict__ alpha,
                       const float* __restrict__ cw,
                       float* __restrict__ convout)
{
  int idx = blockIdx.x * blockDim.x + threadIdx.x;
  if (idx >= MB_ * MT_) return;
  int b = idx / MT_, t = idx % MT_;
  float acc[10];
#pragma unroll
  for (int f = 0; f < 10; ++f) acc[f] = 0.0f;
  const float* ab = alpha + b * MT_;
  for (int k = 0; k < 100; ++k) {
    int src = t - 50 + k;
    if ((unsigned)src < (unsigned)MT_) {
      float a = ab[src];
#pragma unroll
      for (int f = 0; f < 10; ++f) acc[f] += a * cw[f * 100 + k];
    }
  }
#pragma unroll
  for (int f = 0; f < 10; ++f) convout[idx * 10 + f] = acc[f];
}

// energy: e[b][t] = sum_d Wee[d]*tanh(sW[b][d] + hepre[t*16+b][d]
//                                    + sum_f convout[b][t][f]*Wfe[d][f])
__global__ void k_energy(const float* __restrict__ sW,
                         const float* __restrict__ hepre,
                         const float* __restrict__ convout,
                         const float* __restrict__ Wfe,
                         const float* __restrict__ Wee,
                         float* __restrict__ e)
{
  int wid  = blockIdx.x * (blockDim.x >> 5) + (threadIdx.x >> 5);
  int lane = threadIdx.x & 31;
  if (wid >= MB_ * MT_) return;
  int b = wid / MT_, t = wid % MT_;
  float cf[10];
#pragma unroll
  for (int f = 0; f < 10; ++f) cf[f] = convout[wid * 10 + f];
  const float* hp = hepre + ((long long)t * MB_ + b) * 1024;
  const float* sw = sW + b * 1024;
  float acc = 0.0f;
  for (int d = lane; d < 1024; d += 32) {
    float loc = 0.0f;
#pragma unroll
    for (int f = 0; f < 10; ++f) loc += cf[f] * Wfe[d * 10 + f];
    acc += Wee[d] * tanhf(sw[d] + hp[d] + loc);
  }
#pragma unroll
  for (int off = 16; off; off >>= 1) acc += __shfl_xor(acc, off, 32);
  if (lane == 0) e[b * MT_ + t] = acc;
}

// masked softmax over t (max over ALL t, mask applied to exp, as in ref)
__global__ void k_softmax(const float* __restrict__ e,
                          const int* __restrict__ lens,
                          float* __restrict__ alpha)
{
  int b = blockIdx.x;
  int len = lens[b];
  __shared__ float red[256];
  float mx = -1e30f;
  for (int t = threadIdx.x; t < MT_; t += 256) mx = fmaxf(mx, e[b * MT_ + t]);
  red[threadIdx.x] = mx; __syncthreads();
  for (int s = 128; s; s >>= 1) {
    if (threadIdx.x < s) red[threadIdx.x] = fmaxf(red[threadIdx.x], red[threadIdx.x + s]);
    __syncthreads();
  }
  mx = red[0]; __syncthreads();
  float sum = 0.0f;
  for (int t = threadIdx.x; t < MT_; t += 256)
    sum += (t < len) ? __expf(e[b * MT_ + t] - mx) : 0.0f;
  red[threadIdx.x] = sum; __syncthreads();
  for (int s = 128; s; s >>= 1) {
    if (threadIdx.x < s) red[threadIdx.x] += red[threadIdx.x + s];
    __syncthreads();
  }
  float inv = 1.0f / red[0];
  for (int t = threadIdx.x; t < MT_; t += 256)
    alpha[b * MT_ + t] = (t < len) ? __expf(e[b * MT_ + t] - mx) * inv : 0.0f;
}

// context: g[b][d] = sum_t alpha[b][t] * henc[t][b][d]; also bf16 copy
__global__ void k_context(const float* __restrict__ alpha,
                          const float* __restrict__ henc,
                          float* __restrict__ g, bf16* __restrict__ gbf)
{
  int idx = blockIdx.x * blockDim.x + threadIdx.x;
  if (idx >= MB_ * 1024) return;
  int b = idx >> 10, d = idx & 1023;
  const float* ab = alpha + b * MT_;
  float acc = 0.0f;
  for (int t = 0; t < MT_; ++t)
    acc += ab[t] * henc[((long long)t * MB_ + b) * 1024 + d];
  g[idx] = acc;
  gbf[idx] = __float2bfloat16(acc);
}

__global__ void k_tanhc(const float* __restrict__ z, bf16* __restrict__ zt)
{
  int idx = blockIdx.x * blockDim.x + threadIdx.x;
  if (idx >= MB_ * MH_) return;
  zt[idx] = __float2bfloat16(tanhf(z[idx]));
}

// decoder cell: tanh-form sigmoids (0.5*tanh(0.5x)+0.5) per reference
__global__ void k_cell(const float* __restrict__ rec, float* __restrict__ s,
                       float* __restrict__ c, bf16* __restrict__ sbf)
{
  int idx = blockIdx.x * blockDim.x + threadIdx.x;
  if (idx >= MB_ * MH_) return;
  int b = idx >> 9, j = idx & 511;
  const float* rb = rec + b * 2048;
  float iv = tanhf(rb[j]          * 0.5f) * 0.5f + 0.5f;
  float fv = tanhf(rb[512 + j]    * 0.5f) * 0.5f + 0.5f;
  float cg = tanhf(rb[1024 + j]);
  float ov = tanhf(rb[1536 + j]   * 0.5f) * 0.5f + 0.5f;
  float cn = fv * c[idx] + iv * cg;
  float sn = ov * tanhf(cn);
  c[idx] = cn; s[idx] = sn;
  sbf[idx] = __float2bfloat16(sn);
}

// ---------------------------------------------------------------------------
// Host orchestration
// ---------------------------------------------------------------------------
extern "C" void kernel_launch(void* const* d_in, const int* in_sizes, int n_in,
                              void* d_out, int out_size, void* d_ws, size_t ws_size,
                              hipStream_t stream)
{
  (void)in_sizes; (void)n_in; (void)out_size; (void)ws_size;

  const float* x      = (const float*)d_in[0];
  const int*   lens   = (const int*)  d_in[1];
  const float* target = (const float*)d_in[2];
  const float* Wih0   = (const float*)d_in[3];
  const float* Whh0   = (const float*)d_in[4];
  const float* b0     = (const float*)d_in[5];
  const float* WihL   = (const float*)d_in[6];
  const float* WhhL   = (const float*)d_in[7];
  const float* bL     = (const float*)d_in[8];
  const float* W_se   = (const float*)d_in[9];
  const float* W_he   = (const float*)d_in[10];
  const float* b_he   = (const float*)d_in[11];
  const float* W_ee   = (const float*)d_in[12];
  const float* conv_w = (const float*)d_in[13];
  const float* W_fe   = (const float*)d_in[14];
  const float* W_sy   = (const float*)d_in[15];
  const float* W_gy   = (const float*)d_in[16];
  const float* b_gy   = (const float*)d_in[17];
  const float* W_yy   = (const float*)d_in[18];
  const float* b_yy   = (const float*)d_in[19];
  const float* W_ys   = (const float*)d_in[20];
  const float* W_ss   = (const float*)d_in[21];
  const float* W_gs   = (const float*)d_in[22];
  const float* b_gs   = (const float*)d_in[23];

  // ---- workspace layout ----
  char* ws = (char*)d_ws;
  size_t off = 0;
  auto take = [&](size_t bytes) -> char* {
    char* p = ws + off;
    off = (off + bytes + 255) & ~(size_t)255;
    return p;
  };
  bf16* wWih0 = (bf16*)take((size_t)2 * 2048 * 128 * 2);
  bf16* wWihL = (bf16*)take((size_t)6 * 2048 * 1024 * 2);
  bf16* wWhh  = (bf16*)take((size_t)8 * 2048 * 512 * 2);
  bf16* wWse  = (bf16*)take((size_t)1024 * 512 * 2);
  bf16* wWhe  = (bf16*)take((size_t)1024 * 1024 * 2);
  bf16* wWys  = (bf16*)take((size_t)2048 * KC_ * 2);
  bf16* wWss  = (bf16*)take((size_t)2048 * 512 * 2);
  bf16* wWgs  = (bf16*)take((size_t)2048 * 1024 * 2);
  bf16* wWgy  = (bf16*)take((size_t)512 * 1024 * 2);
  bf16* wWsy  = (bf16*)take((size_t)512 * 512 * 2);
  bf16* wWyy  = (bf16*)take((size_t)5008 * 512 * 2);
  bf16*  Abuf  = (bf16*) take((size_t)MT_ * MB_ * 1024 * 2);
  float* xpF   = (float*)take((size_t)MT_ * MB_ * 2048 * 4);
  float* xpB   = (float*)take((size_t)MT_ * MB_ * 2048 * 4);
  float* hencA = (float*)take((size_t)MT_ * MB_ * 1024 * 4);
  float* hencB = (float*)take((size_t)MT_ * MB_ * 1024 * 4);
  float* alpha = (float*)take((size_t)MB_ * MT_ * 4);
  float* ebuf  = (float*)take((size_t)MB_ * MT_ * 4);
  float* convo = (float*)take((size_t)MB_ * MT_ * 10 * 4);
  float* sWb   = (float*)take((size_t)MB_ * 1024 * 4);
  float* gctx  = (float*)take((size_t)MB_ * 1024 * 4);
  bf16*  gbf   = (bf16*) take((size_t)MB_ * 1024 * 2);
  float* sdec  = (float*)take((size_t)MB_ * MH_ * 4);
  float* cdec  = (float*)take((size_t)MB_ * MH_ * 4);
  bf16*  sbf   = (bf16*) take((size_t)MB_ * MH_ * 2);
  float* zbuf  = (float*)take((size_t)MB_ * MH_ * 4);
  bf16*  ztbf  = (bf16*) take((size_t)MB_ * MH_ * 2);
  float* rec   = (float*)take((size_t)MB_ * 2048 * 4);

  auto cvt = [&](const float* src, int sR, int sC, bf16* dst, int dR, int dC) {
    long long tot = (long long)dR * dC;
    int blocks = (int)((tot + 255) / 256);
    k_cvt<<<blocks, 256, 0, stream>>>(src, sR, sC, dst, dR, dC);
  };
  auto gemm = [&](const bf16* A, int lda, const bf16* W, int ldw,
                  float* Cp, long long ldc, const float* bias,
                  const float* addsrc, long long ldadd,
                  int M, int N, int Npad16, int K, int accum) {
    int n64 = (Npad16 + 63) >> 6;
    int waves = ((M + 15) >> 4) * n64;
    int blocks = (waves + 7) >> 3;
    k_gemm<<<blocks, 256, 0, stream>>>(A, lda, W, ldw, Cp, ldc, bias,
                                       addsrc, ldadd, M, N, Npad16, K, accum);
  };

  // ---- weight conversions (bf16, K padded to %64, N padded to %16) ----
  for (int d = 0; d < 2; ++d) {
    cvt(Wih0 + (size_t)d * 2048 * MF_, 2048, MF_, wWih0 + (size_t)d * 2048 * 128, 2048, 128);
    cvt(Whh0 + (size_t)d * 2048 * 512, 2048, 512, wWhh  + (size_t)d * 2048 * 512, 2048, 512);
  }
  for (int l = 1; l < 4; ++l)
    for (int d = 0; d < 2; ++d) {
      size_t s2 = (size_t)((l - 1) * 2 + d);
      cvt(WihL + s2 * 2048 * 1024, 2048, 1024, wWihL + s2 * 2048 * 1024, 2048, 1024);
      cvt(WhhL + s2 * 2048 * 512,  2048, 512,  wWhh + (size_t)(l * 2 + d) * 2048 * 512, 2048, 512);
    }
  cvt(W_se, 1024, 512,  wWse, 1024, 512);
  cvt(W_he, 1024, 1024, wWhe, 1024, 1024);
  cvt(W_ys, 2048, MC_,  wWys, 2048, KC_);
  cvt(W_ss, 2048, 512,  wWss, 2048, 512);
  cvt(W_gs, 2048, 1024, wWgs, 2048, 1024);
  cvt(W_gy, 512, 1024,  wWgy, 512, 1024);
  cvt(W_sy, 512, 512,   wWsy, 512, 512);
  cvt(W_yy, MC_, 512,   wWyy, 5008, 512);

  // ---- encoder ----
  // layer 0: x projections (hoisted out of the recurrence), then persistent LSTM
  k_cvt_x<<<(MT_ * MB_ * 128 + 255) / 256, 256, 0, stream>>>(x, Abuf);
  gemm(Abuf, 128, wWih0,                    128, xpF, 2048, b0,        nullptr, 0,
       MT_ * MB_, 2048, 2048, 128, 0);
  gemm(Abuf, 128, wWih0 + (size_t)2048*128, 128, xpB, 2048, b0 + 2048, nullptr, 0,
       MT_ * MB_, 2048, 2048, 128, 0);
  k_lstm<<<2, 512, 0, stream>>>(wWhh, wWhh + (size_t)2048 * 512, xpF, xpB, lens, hencA);

  float* lin = hencA;
  float* lout = hencB;
  for (int l = 1; l < 4; ++l) {
    cvt(lin, MT_ * MB_, 1024, Abuf, MT_ * MB_, 1024);
    size_t i0 = (size_t)((l - 1) * 2 + 0), i1 = (size_t)((l - 1) * 2 + 1);
    gemm(Abuf, 1024, wWihL + i0 * 2048 * 1024, 1024, xpF, 2048, bL + i0 * 2048,
         nullptr, 0, MT_ * MB_, 2048, 2048, 1024, 0);
    gemm(Abuf, 1024, wWihL + i1 * 2048 * 1024, 1024, xpB, 2048, bL + i1 * 2048,
         nullptr, 0, MT_ * MB_, 2048, 2048, 1024, 0);
    k_lstm<<<2, 512, 0, stream>>>(wWhh + (size_t)(l * 2 + 0) * 2048 * 512,
                                  wWhh + (size_t)(l * 2 + 1) * 2048 * 512,
                                  xpF, xpB, lens, lout);
    float* tmp = lin; lin = lout; lout = tmp;
  }
  float* henc  = lin;    // final encoder output [T][16][1024]
  float* hepre = lout;   // reuse the other ping buffer for he_pre

  // hoisted decoder-invariant GEMMs
  cvt(henc, MT_ * MB_, 1024, Abuf, MT_ * MB_, 1024);   // hbatch in bf16
  gemm(Abuf, 1024, wWhe, 1024, hepre, 1024, b_he, nullptr, 0,
       MT_ * MB_, 1024, 1024, 1024, 0);
  bf16*  tgtbf = (bf16*)xpF;   // xproj buffers are free after the encoder
  float* yspre = xpB;
  cvt(target, MB_ * MNL_, MC_, tgtbf, MB_ * MNL_, KC_);
  gemm(tgtbf, KC_, wWys, KC_, yspre, 2048, b_gs, nullptr, 0,
       MB_ * MNL_, 2048, 2048, KC_, 0);

  // ---- decoder ----
  hipMemsetAsync(alpha, 0, (size_t)MB_ * MT_ * 4, stream);
  hipMemsetAsync(sdec,  0, (size_t)MB_ * MH_ * 4, stream);
  hipMemsetAsync(cdec,  0, (size_t)MB_ * MH_ * 4, stream);
  hipMemsetAsync(sbf,   0, (size_t)MB_ * MH_ * 2, stream);

  for (int step = 0; step < MNL_; ++step) {
    k_conv<<<(MB_ * MT_ + 255) / 256, 256, 0, stream>>>(alpha, conv_w, convo);
    gemm(sbf, 512, wWse, 512, sWb, 1024, nullptr, nullptr, 0,
         MB_, 1024, 1024, 512, 0);
    k_energy<<<(MB_ * MT_ + 7) / 8, 256, 0, stream>>>(sWb, hepre, convo, W_fe, W_ee, ebuf);
    k_softmax<<<MB_, 256, 0, stream>>>(ebuf, lens, alpha);
    k_context<<<(MB_ * 1024 + 255) / 256, 256, 0, stream>>>(alpha, henc, gctx, gbf);
    // z = g@W_gy^T + b_gy + s@W_sy^T ; y = tanh(z)@W_yy^T + b_yy -> d_out
    gemm(gbf, 1024, wWgy, 1024, zbuf, 512, b_gy, nullptr, 0, MB_, 512, 512, 1024, 0);
    gemm(sbf, 512,  wWsy, 512,  zbuf, 512, nullptr, nullptr, 0, MB_, 512, 512, 512, 1);
    k_tanhc<<<(MB_ * MH_ + 255) / 256, 256, 0, stream>>>(zbuf, ztbf);
    gemm(ztbf, 512, wWyy, 512, (float*)d_out + (size_t)step * MC_,
         (long long)MNL_ * MC_, b_yy, nullptr, 0, MB_, MC_, 5008, 512, 0);
    // rec = ys_pre[step] + s@W_ss^T + g@W_gs^T (b_gs already folded into ys_pre)
    gemm(sbf, 512,  wWss, 512,  rec, 2048, nullptr, yspre + (size_t)step * 2048,
         (long long)MNL_ * 2048, MB_, 2048, 2048, 512, 0);
    gemm(gbf, 1024, wWgs, 1024, rec, 2048, nullptr, nullptr, 0,
         MB_, 2048, 2048, 1024, 1);
    k_cell<<<(MB_ * MH_ + 255) / 256, 256, 0, stream>>>(rec, sdec, cdec, sbf);
  }
}
